// LayerNormLSTM_43164421325554
// MI455X (gfx1250) — matrix operand order repaired and checked by reading, not compiled, not run on hardware
//
#include <hip/hip_runtime.h>
#include <hip/hip_bf16.h>
#include <math.h>

typedef __attribute__((ext_vector_type(16))) __bf16 v16bf;
typedef __attribute__((ext_vector_type(8)))  float  v8f;

#define T_STEPS 512
#define BATCH   128
#define IDIM    512
#define HDIM    512
#define GDIM    2048   // 4*H
#define KB_W    16     // IDIM/32 k-blocks
#define LN_EPS  1e-5f

union FragBF {
    v16bf v;
    uint4 q[2];
};

// ---------------------------------------------------------------------------
// Prologue: pack W,R (fp32 row-major KxN) into bf16 WMMA-B lane layout:
//   Wp[((kb*GDIM + n)*32) + j] = W[(kb*32+j)*GDIM + n]
// (lane = n; each lane-half reads 16 contiguous bf16 -> contiguous 64B groups)
// Also zero h (bf16) and c (f32) state.
// ---------------------------------------------------------------------------
__global__ void lnlstm_init(const float* __restrict__ W, const float* __restrict__ R,
                            __hip_bfloat16* __restrict__ Wp, __hip_bfloat16* __restrict__ Rp,
                            __hip_bfloat16* __restrict__ h0, float* __restrict__ c0) {
    size_t idx = (size_t)blockIdx.x * blockDim.x + threadIdx.x;
    const size_t NPACK = (size_t)GDIM * IDIM;   // 1,048,576
    if (idx < NPACK) {
        int j  = (int)(idx & 31);
        int n  = (int)((idx >> 5) & (GDIM - 1));
        int kb = (int)(idx >> 16);              // 2048*32 = 65536 per k-block
        int k  = kb * 32 + j;
        Wp[idx] = __float2bfloat16(W[(size_t)k * GDIM + n]);
        Rp[idx] = __float2bfloat16(R[(size_t)k * GDIM + n]);
    }
    if (idx < (size_t)BATCH * HDIM) {
        h0[idx] = __float2bfloat16(0.0f);
        c0[idx] = 0.0f;
    }
}

// ---------------------------------------------------------------------------
// One LSTM timestep. grid = B/16 blocks; block = 1024 threads (32 waves).
// Each block owns 16 batch rows and ALL 2048 gate columns, so every LN row
// reduction is workgroup-local (LDS ds_add_f32 atomics).
// Two-phase accumulation (Wx then Rh) halves live accumulator VGPRs.
// ---------------------------------------------------------------------------
__global__ __launch_bounds__(1024, 1)
void lnlstm_step(const float* __restrict__ x,              // (T,B,I)
                 const __hip_bfloat16* __restrict__ Wp,    // packed bf16
                 const __hip_bfloat16* __restrict__ Rp,    // packed bf16
                 const float* __restrict__ bias,           // (4H)
                 const float* __restrict__ gamma,          // (2,4H)
                 const float* __restrict__ gamma_h,        // (H)
                 const float* __restrict__ beta_h,         // (H)
                 __hip_bfloat16* __restrict__ hbf,         // (B,H) state, bf16
                 float* __restrict__ cst,                  // (B,H) state, f32
                 float* __restrict__ out,                  // hs | h_n | c_n
                 int t) {
    // Non-aliased LDS: A stays valid through GEMM 2 while vbuf accumulates.
    __shared__ __align__(16) __hip_bfloat16 Alds[16 * 1024];  // 32 KB: [x_t | h]
    __shared__ __align__(16) float vbuf[16 * GDIM];           // 128 KB pre-activations
    __shared__ float rs[2][16], rss[2][16];   // Wx / Rh row stats
    __shared__ float cs[16], css[16];         // c row stats

    const int tid     = threadIdx.x;
    const int rowbase = blockIdx.x * 16;

    if (tid < 16) {
        rs[0][tid] = 0.f; rs[1][tid] = 0.f;
        rss[0][tid] = 0.f; rss[1][tid] = 0.f;
        cs[tid] = 0.f; css[tid] = 0.f;
    }

    // ---- stage A = [x_t (16x512) | h (16x512)] in LDS as bf16 ----
    for (int k = 0; k < 8; ++k) {
        int idx = tid + (k << 10);
        int r = idx >> 9, i = idx & 511;
        Alds[r * 1024 + i]       = __float2bfloat16(x[((size_t)t * BATCH + rowbase + r) * IDIM + i]);
        Alds[r * 1024 + 512 + i] = hbf[(size_t)(rowbase + r) * HDIM + i];
    }
    __syncthreads();

    const int wave  = tid >> 5;
    const int lane  = tid & 31;
    const int m     = lane & 15;     // A row / B column-in-tile
    const int khalf = lane >> 4;     // which K half this lane holds
    const int n0    = wave << 2;     // 4 N-tiles per wave
    const float inv_g = 1.0f / (float)GDIM;

    v8f zacc = {};
    v8f acc[4];

    // ================= Phase 1: Wx = x_t @ W =================
    for (int u = 0; u < 4; ++u) acc[u] = zacc;
    for (int kb = 0; kb < KB_W; ++kb) {
        FragBF a;
        const __hip_bfloat16* ap = Alds + m * 1024 + kb * 32 + (khalf << 4);
        a.q[0] = *(const uint4*)(ap);
        a.q[1] = *(const uint4*)(ap + 8);
        if (kb + 2 < KB_W) {  // cover first-touch latency of packed weights
            __builtin_prefetch(Wp + (((size_t)(kb + 2) * GDIM + (n0 << 4) + m) << 5), 0, 0);
        }
        for (int u = 0; u < 4; ++u) {
            int ncol = ((n0 + u) << 4) + m;
            const __hip_bfloat16* bp = Wp + (((size_t)kb * GDIM + ncol) << 5) + (khalf << 4);
            FragBF b;
            b.q[0] = *(const uint4*)(bp);
            b.q[1] = *(const uint4*)(bp + 8);
            acc[u] = __builtin_amdgcn_wmma_f32_16x16x32_bf16(
                false, a.v, false, b.v, (short)0, acc[u], false, false);
        }
    }
    // row stats over 2048 cols; acc element r -> row (r + khalf*8)
    for (int r = 0; r < 8; ++r) {
        int mrow = r + (khalf << 3);
        float s = 0.f, ss = 0.f;
        for (int u = 0; u < 4; ++u) { float a = acc[u][r]; s += a; ss += a * a; }
        atomicAdd(&rs[0][mrow], s);
        atomicAdd(&rss[0][mrow], ss);
    }
    __syncthreads();
    // vbuf = LN(Wx)*g0 + bias   (this lane owns these (row,col) slots)
    for (int r = 0; r < 8; ++r) {
        int mrow = r + (khalf << 3);
        float mwx = rs[0][mrow] * inv_g;
        float rwx = rsqrtf(rss[0][mrow] * inv_g - mwx * mwx + LN_EPS);
        for (int u = 0; u < 4; ++u) {
            int ncol = ((n0 + u) << 4) + m;
            vbuf[mrow * GDIM + ncol] = (acc[u][r] - mwx) * rwx * gamma[ncol] + bias[ncol];
        }
    }

    // ================= Phase 2: Rh = h @ R =================
    for (int u = 0; u < 4; ++u) acc[u] = zacc;
    for (int kb = 0; kb < KB_W; ++kb) {
        FragBF a;
        const __hip_bfloat16* ap = Alds + m * 1024 + 512 + kb * 32 + (khalf << 4);
        a.q[0] = *(const uint4*)(ap);
        a.q[1] = *(const uint4*)(ap + 8);
        if (kb + 2 < KB_W) {
            __builtin_prefetch(Rp + (((size_t)(kb + 2) * GDIM + (n0 << 4) + m) << 5), 0, 0);
        }
        for (int u = 0; u < 4; ++u) {
            int ncol = ((n0 + u) << 4) + m;
            const __hip_bfloat16* bp = Rp + (((size_t)kb * GDIM + ncol) << 5) + (khalf << 4);
            FragBF b;
            b.q[0] = *(const uint4*)(bp);
            b.q[1] = *(const uint4*)(bp + 8);
            acc[u] = __builtin_amdgcn_wmma_f32_16x16x32_bf16(
                false, a.v, false, b.v, (short)0, acc[u], false, false);
        }
    }
    for (int r = 0; r < 8; ++r) {
        int mrow = r + (khalf << 3);
        float s = 0.f, ss = 0.f;
        for (int u = 0; u < 4; ++u) { float a = acc[u][r]; s += a; ss += a * a; }
        atomicAdd(&rs[1][mrow], s);
        atomicAdd(&rss[1][mrow], ss);
    }
    __syncthreads();
    // vbuf += LN(Rh)*g1  (same-thread RMW on owned slots -> race-free)
    for (int r = 0; r < 8; ++r) {
        int mrow = r + (khalf << 3);
        float mrh = rs[1][mrow] * inv_g;
        float rrh = rsqrtf(rss[1][mrow] * inv_g - mrh * mrh + LN_EPS);
        for (int u = 0; u < 4; ++u) {
            int ncol = ((n0 + u) << 4) + m;
            vbuf[mrow * GDIM + ncol] += (acc[u][r] - mrh) * rrh * gamma[GDIM + ncol];
        }
    }
    __syncthreads();

    // ---- gates + c update + LN(c) stats ----
    float cnv[8], ov[8];
    for (int k = 0; k < 8; ++k) {
        int idx = tid + (k << 10);
        int r = idx >> 9, hc = idx & 511;
        float vi = vbuf[r * GDIM + hc];
        float vg = vbuf[r * GDIM + 512 + hc];
        float vf = vbuf[r * GDIM + 1024 + hc];
        float vo = vbuf[r * GDIM + 1536 + hc];
        float ig = 1.0f / (1.0f + __expf(-vi));
        float gg = tanhf(vg);
        float fg = 1.0f / (1.0f + __expf(-vf));
        float og = 1.0f / (1.0f + __expf(-vo));
        size_t sidx = (size_t)(rowbase + r) * HDIM + hc;
        float cn = fg * cst[sidx] + ig * gg;
        cst[sidx] = cn;
        cnv[k] = cn; ov[k] = og;
        atomicAdd(&cs[r], cn);
        atomicAdd(&css[r], cn * cn);
    }
    __syncthreads();

    // ---- h = o * tanh(LN(c)*gh + bh); write outputs + next state ----
    const float inv_h = 1.0f / (float)HDIM;
    for (int k = 0; k < 8; ++k) {
        int idx = tid + (k << 10);
        int r = idx >> 9, hc = idx & 511;
        float mc = cs[r] * inv_h;
        float rc = rsqrtf(css[r] * inv_h - mc * mc + LN_EPS);
        float cnorm = (cnv[k] - mc) * rc * gamma_h[hc] + beta_h[hc];
        float hn = ov[k] * tanhf(cnorm);
        size_t sidx = (size_t)(rowbase + r) * HDIM + hc;
        out[(size_t)t * BATCH * HDIM + sidx] = hn;
        hbf[sidx] = __float2bfloat16(hn);
        if (t == T_STEPS - 1) {
            out[(size_t)T_STEPS * BATCH * HDIM + sidx] = hn;                            // h_n
            out[(size_t)T_STEPS * BATCH * HDIM + (size_t)BATCH * HDIM + sidx] = cnv[k]; // c_n
        }
    }
}

extern "C" void kernel_launch(void* const* d_in, const int* in_sizes, int n_in,
                              void* d_out, int out_size, void* d_ws, size_t ws_size,
                              hipStream_t stream) {
    const float* x       = (const float*)d_in[0];
    const float* W       = (const float*)d_in[1];
    const float* R       = (const float*)d_in[2];
    const float* bias    = (const float*)d_in[3];
    const float* gamma   = (const float*)d_in[4];
    const float* gamma_h = (const float*)d_in[5];
    const float* beta_h  = (const float*)d_in[6];
    float* out = (float*)d_out;

    char* ws = (char*)d_ws;
    size_t packBytes = (size_t)GDIM * IDIM * sizeof(__hip_bfloat16);   // 2 MB
    __hip_bfloat16* Wp  = (__hip_bfloat16*)(ws);
    __hip_bfloat16* Rp  = (__hip_bfloat16*)(ws + packBytes);
    __hip_bfloat16* hbf = (__hip_bfloat16*)(ws + 2 * packBytes);
    float*          cst = (float*)(ws + 2 * packBytes + (size_t)BATCH * HDIM * sizeof(__hip_bfloat16));

    size_t npack = (size_t)GDIM * IDIM;
    int initBlocks = (int)((npack + 255) / 256);
    lnlstm_init<<<initBlocks, 256, 0, stream>>>(W, R, Wp, Rp, hbf, cst);

    for (int t = 0; t < T_STEPS; ++t) {
        lnlstm_step<<<BATCH / 16, 1024, 0, stream>>>(
            x, Wp, Rp, bias, gamma, gamma_h, beta_h, hbf, cst, out, t);
    }
}
